// PointGenerator_16140487098442
// MI455X (gfx1250) — matrix operand order repaired and checked by reading, not compiled, not run on hardware
//
#include <hip/hip_runtime.h>
#include <hip/hip_bf16.h>

typedef __attribute__((ext_vector_type(16))) __bf16 v16bf;
typedef __attribute__((ext_vector_type(8)))  __bf16 bf16x8;
typedef __attribute__((ext_vector_type(8)))  float  v8f;

#define DEVINL __device__ __forceinline__

DEVINL v8f wmma_bf16(v16bf a, v16bf b, v8f c) {
  // D = A(16x32 bf16) * B(32x16 bf16) + C(16x16 f32)
  return __builtin_amdgcn_wmma_f32_16x16x32_bf16(false, a, false, b, (short)0, c,
                                                 false, false);
}

DEVINL v16bf cat8(bf16x8 a, bf16x8 b) {
  return __builtin_shufflevector(a, b, 0, 1, 2, 3, 4, 5, 6, 7,
                                 8, 9, 10, 11, 12, 13, 14, 15);
}
DEVINL bf16x8 ld8(const __bf16* __restrict__ p) { return *(const bf16x8*)p; }

// CDNA5 16-bit A 16x32 layout: lane holds row (l&15); half-wave 0 K={0..7,16..23},
// half-wave 1 K={8..15,24..31}.  B 32x16: lane holds col; same K split as A.
// All K-strides are padded to %32 elements -> unconditional 16B loads, no guards.
template<int K>
DEVINL v16bf load_afrag(const __bf16* __restrict__ row, int kt, int grp) {
  static_assert(K % 32 == 0, "A stride must be padded to 32");
  int kb = (kt << 5) + (grp << 3);
  return cat8(ld8(row + kb), ld8(row + kb + 16));
}
template<int K>
DEVINL v16bf load_bfrag(const __bf16* __restrict__ colp, int kt, int grp) {
  static_assert(K % 32 == 0, "B stride must be padded to 32");
  int kb = (kt << 5) + (grp << 3);
  return cat8(ld8(colp + kb), ld8(colp + kb + 16));
}
// Edge row [xi(CP) | xj - xi(CP)]: branch depends only on kt (folds at unroll).
template<int CP>
DEVINL v16bf load_edge_afrag(const __bf16* __restrict__ Xi, const __bf16* __restrict__ Xj,
                             int kt, int grp) {
  static_assert(CP % 32 == 0, "edge feature stride must be padded to 32");
  int kb = (kt << 5) + (grp << 3);
  if (kt < CP / 32) {
    return cat8(ld8(Xi + kb), ld8(Xi + kb + 16));
  } else {
    bf16x8 d0 = ld8(Xj + kb - CP) - ld8(Xi + kb - CP);
    bf16x8 d1 = ld8(Xj + kb - CP + 16) - ld8(Xi + kb - CP + 16);
    return cat8(d0, d1);
  }
}

// ---------------------------------------------------------------------------
// Fused pairwise-distance + top-k kNN. One wave per 16-query tile; query A
// fragments cached in VGPRs across the whole column sweep.
// ---------------------------------------------------------------------------
template<int D, int KNBR>
__global__ void __launch_bounds__(32) knn_kernel(const __bf16* __restrict__ x,
                                                 const float* __restrict__ sq,
                                                 int N, int* __restrict__ nbr) {
  __shared__ float dtile[16][17];
  const int lane = threadIdx.x;
  const int mlo = lane & 15, grp = lane >> 4;
  const int row0 = blockIdx.x << 4;
  constexpr int KT = D / 32;

  const __bf16* arow = x + (size_t)(row0 + mlo) * D;
  v16bf afrag[KT];
#pragma unroll
  for (int kt = 0; kt < KT; ++kt) afrag[kt] = load_afrag<D>(arow, kt, grp);

  float bestd[KNBR]; int besti[KNBR];
#pragma unroll
  for (int q = 0; q < KNBR; ++q) { bestd[q] = 3.0e38f; besti[q] = 0; }

  const int ntiles = N >> 4;
  for (int jt = 0; jt < ntiles; ++jt) {
    const int col0 = jt << 4;
    const __bf16* brow = x + (size_t)(col0 + mlo) * D;
    v8f acc;
#pragma unroll
    for (int r = 0; r < 8; ++r) acc[r] = 0.0f;
#pragma unroll
    for (int kt = 0; kt < KT; ++kt)
      acc = wmma_bf16(afrag[kt], load_bfrag<D>(brow, kt, grp), acc);
    const float sqc = sq[col0 + mlo];
#pragma unroll
    for (int r = 0; r < 8; ++r) {
      int m = (grp << 3) + r;
      float d = sq[row0 + m] + sqc - 2.0f * acc[r];
      if (row0 + m == col0 + mlo) d += 1.0e10f;   // mask self
      dtile[m][mlo] = d;
    }
    __syncthreads();
    if (lane < 16) {
#pragma unroll
      for (int n = 0; n < 16; ++n) {
        float d = dtile[lane][n];
        if (d < bestd[KNBR - 1]) {               // threshold-reject fast path
          bestd[KNBR - 1] = d; besti[KNBR - 1] = col0 + n;
#pragma unroll
          for (int q = KNBR - 1; q > 0; --q) {   // single insert pass
            if (bestd[q] < bestd[q - 1]) {
              float td = bestd[q]; bestd[q] = bestd[q - 1]; bestd[q - 1] = td;
              int ti = besti[q]; besti[q] = besti[q - 1]; besti[q - 1] = ti;
            }
          }
        }
      }
    }
    __syncthreads();
  }
  if (lane < 16) {
#pragma unroll
    for (int q = 0; q < KNBR; ++q) nbr[(row0 + lane) * KNBR + q] = besti[q];
  }
}

// ---------------------------------------------------------------------------
// Generic WMMA GEMM: out = resid + scale * act(A(MxK) @ W(KxN) + bias).
// A, Wt in bf16 (Wt pre-transposed NxK, K padded). OutT = bf16 intermediates
// or float finals. LDO = padded out stride; pad columns zero-filled.
// ---------------------------------------------------------------------------
template<int K, int N, int LDO, int NT, typename OutT>
__global__ void __launch_bounds__(32) gemm_kernel(
    const __bf16* __restrict__ A, const __bf16* __restrict__ Wt,
    const float* __restrict__ bias, const float* __restrict__ resid,
    OutT* __restrict__ out, int relu, float scale) {
  const int lane = threadIdx.x;
  const int mlo = lane & 15, grp = lane >> 4;
  const int row0 = blockIdx.x << 4;
  const int n0 = (blockIdx.y * NT) << 4;
  const __bf16* arow = A + (size_t)(row0 + mlo) * K;
  const __bf16* bcol[NT];
#pragma unroll
  for (int t = 0; t < NT; ++t) {
    int c = n0 + t * 16 + mlo;
    if (c > N - 1) c = N - 1;            // clamp: dup column, store is masked
    bcol[t] = Wt + (size_t)c * K;
  }
  v8f acc[NT];
#pragma unroll
  for (int t = 0; t < NT; ++t)
#pragma unroll
    for (int r = 0; r < 8; ++r) acc[t][r] = 0.0f;
  constexpr int KT = K / 32;
#pragma unroll
  for (int kt = 0; kt < KT; ++kt) {
    v16bf af = load_afrag<K>(arow, kt, grp);
#pragma unroll
    for (int t = 0; t < NT; ++t)
      acc[t] = wmma_bf16(af, load_bfrag<K>(bcol[t], kt, grp), acc[t]);
  }
#pragma unroll
  for (int t = 0; t < NT; ++t) {
    int col = n0 + t * 16 + mlo;
    if (col < N) {
      float b = bias ? bias[col] : 0.0f;
#pragma unroll
      for (int r = 0; r < 8; ++r) {
        int m = row0 + (grp << 3) + r;
        float v = acc[t][r] + b;
        if (relu) v = fmaxf(v, 0.0f);
        v *= scale;
        if (resid) v += resid[(size_t)m * LDO + col];
        out[(size_t)m * LDO + col] = (OutT)v;
      }
    } else if (col < LDO) {              // zero-fill pad columns
#pragma unroll
      for (int r = 0; r < 8; ++r)
        out[(size_t)(row0 + (grp << 3) + r) * LDO + col] = (OutT)0.0f;
    }
  }
}

// ---------------------------------------------------------------------------
// Edge-conv layer 1: rows are edges; A[e] = [x_i | x_j - x_i] gathered on the
// fly (each half CP-padded, bf16). out = relu(A @ W + bias), bf16 per edge.
// ---------------------------------------------------------------------------
template<int KNBR, int CP, int N, int NT>
__global__ void __launch_bounds__(32) gemm_edge_kernel(
    const __bf16* __restrict__ X, const int* __restrict__ nbr,
    const __bf16* __restrict__ Wt, const float* __restrict__ bias,
    __bf16* __restrict__ out) {
  constexpr int K = 2 * CP;
  const int lane = threadIdx.x;
  const int mlo = lane & 15, grp = lane >> 4;
  const int row0 = blockIdx.x << 4;
  const int n0 = (blockIdx.y * NT) << 4;
  const int e = row0 + mlo;
  const __bf16* Xi = X + (size_t)(e / KNBR) * CP;
  const __bf16* Xj = X + (size_t)nbr[e] * CP;
  const __bf16* bcol[NT];
#pragma unroll
  for (int t = 0; t < NT; ++t) {
    int c = n0 + t * 16 + mlo;
    if (c > N - 1) c = N - 1;
    bcol[t] = Wt + (size_t)c * K;
  }
  v8f acc[NT];
#pragma unroll
  for (int t = 0; t < NT; ++t)
#pragma unroll
    for (int r = 0; r < 8; ++r) acc[t][r] = 0.0f;
  constexpr int KT = K / 32;
#pragma unroll
  for (int kt = 0; kt < KT; ++kt) {
    v16bf af = load_edge_afrag<CP>(Xi, Xj, kt, grp);
#pragma unroll
    for (int t = 0; t < NT; ++t)
      acc[t] = wmma_bf16(af, load_bfrag<K>(bcol[t], kt, grp), acc[t]);
  }
#pragma unroll
  for (int t = 0; t < NT; ++t) {
    int col = n0 + t * 16 + mlo;
    if (col < N) {
      float b = bias[col];
#pragma unroll
      for (int r = 0; r < 8; ++r) {
        int m = row0 + (grp << 3) + r;
        out[(size_t)m * N + col] = (__bf16)fmaxf(acc[t][r] + b, 0.0f);
      }
    }
  }
}

// ---------------------------------------------------------------------------
// Edge-conv layer 2: GEMM over edge-hidden rows with fused max-over-neighbor
// group + bias + optional residual. KNBR=16: 1 point per tile (combine halves
// via shfl_xor 16 under wave32); KNBR=8: 2 points per tile.
// ---------------------------------------------------------------------------
template<int KNBR, int K, int N, int NT, typename OutT>
__global__ void __launch_bounds__(32) gemm_groupmax_kernel(
    const __bf16* __restrict__ H, const __bf16* __restrict__ Wt,
    const float* __restrict__ bias, const float* __restrict__ resid,
    OutT* __restrict__ out) {
  const int lane = threadIdx.x;
  const int mlo = lane & 15, grp = lane >> 4;
  const int row0 = blockIdx.x << 4;     // edge row base
  const int n0 = (blockIdx.y * NT) << 4;
  const __bf16* arow = H + (size_t)(row0 + mlo) * K;
  const __bf16* bcol[NT];
#pragma unroll
  for (int t = 0; t < NT; ++t) {
    int c = n0 + t * 16 + mlo;
    if (c > N - 1) c = N - 1;
    bcol[t] = Wt + (size_t)c * K;
  }
  v8f acc[NT];
#pragma unroll
  for (int t = 0; t < NT; ++t)
#pragma unroll
    for (int r = 0; r < 8; ++r) acc[t][r] = 0.0f;
  constexpr int KT = K / 32;
#pragma unroll
  for (int kt = 0; kt < KT; ++kt) {
    v16bf af = load_afrag<K>(arow, kt, grp);
#pragma unroll
    for (int t = 0; t < NT; ++t)
      acc[t] = wmma_bf16(af, load_bfrag<K>(bcol[t], kt, grp), acc[t]);
  }
#pragma unroll
  for (int t = 0; t < NT; ++t) {
    int col = n0 + t * 16 + mlo;
    float vmax = acc[t][0];
#pragma unroll
    for (int r = 1; r < 8; ++r) vmax = fmaxf(vmax, acc[t][r]);
    if (KNBR == 16) {
      float o = __shfl_xor(vmax, 16, 32);
      vmax = fmaxf(vmax, o);
      if (grp == 0 && col < N) {
        int pt = row0 >> 4;
        float v = vmax + bias[col];
        if (resid) v += resid[(size_t)pt * N + col];
        out[(size_t)pt * N + col] = (OutT)v;
      }
    } else {
      if (col < N) {
        int pt = (row0 >> 3) + grp;
        float v = vmax + bias[col];
        if (resid) v += resid[(size_t)pt * N + col];
        out[(size_t)pt * N + col] = (OutT)v;
      }
    }
  }
}

// ------------------------- small elementwise kernels -----------------------
// Wt[n*KP + k] = bf16( (k < K) ? W[k*N + n] : 0 )
__global__ void transpose_pad_kernel(const float* __restrict__ W, __bf16* __restrict__ Wt,
                                     int K, int N, int KP) {
  int t = blockIdx.x * blockDim.x + threadIdx.x;
  if (t >= N * KP) return;
  int n = t / KP, k = t % KP;
  Wt[t] = (__bf16)((k < K) ? W[(size_t)k * N + n] : 0.0f);
}

// Edge weight: original rows [0,C) -> [0,C), rows [C,2C) -> [CP, CP+C); pad 0.
__global__ void transpose_edge_pad_kernel(const float* __restrict__ W,
                                          __bf16* __restrict__ Wt,
                                          int C, int CP, int N) {
  int t = blockIdx.x * blockDim.x + threadIdx.x;
  if (t >= N * 2 * CP) return;
  int n = t / (2 * CP), pos = t % (2 * CP);
  int k = (pos < CP) ? ((pos < C) ? pos : -1)
                     : ((pos - CP < C) ? C + (pos - CP) : -1);
  Wt[t] = (__bf16)((k >= 0) ? W[(size_t)k * N + n] : 0.0f);
}

__global__ void f32_to_bf16_kernel(const float* __restrict__ src,
                                   __bf16* __restrict__ dst, int n) {
  int t = blockIdx.x * blockDim.x + threadIdx.x;
  if (t < n) dst[t] = (__bf16)src[t];
}

__global__ void gather_pred_kernel(const float* __restrict__ pred,
                                   const int* __restrict__ mask,
                                   __bf16* __restrict__ out) {
  int t = blockIdx.x * blockDim.x + threadIdx.x;
  if (t >= 2048 * 384) return;
  int p = t / 384, c = t % 384;
  int b = p >> 9, pp = p & 511;
  int m = mask[0];
  out[t] = (__bf16)pred[(size_t)(((b * 4 + m) << 9) + pp) * 384 + c];
}

__global__ void repeat4_kernel(const __bf16* __restrict__ src, __bf16* __restrict__ dst) {
  int t = blockIdx.x * blockDim.x + threadIdx.x;
  if (t >= 8192 * 384) return;
  int r = t / 384, c = t % 384;
  dst[t] = src[(size_t)(r >> 2) * 384 + c];
}

__global__ void tgtxyz_kernel(const float* __restrict__ seed,
                              const float* __restrict__ noise,
                              float* __restrict__ out) {
  int t = blockIdx.x * blockDim.x + threadIdx.x;
  if (t >= 8192 * 3) return;
  int row = t / 3, c = t % 3;
  int b = row >> 11, q = row & 2047;
  out[t] = seed[(size_t)(((b << 9) + (q >> 2)) * 3 + c)] + noise[t];
}

__global__ void sqnorm_kernel(const __bf16* __restrict__ x, int N, int D,
                              float* __restrict__ sq) {
  int p = blockIdx.x * blockDim.x + threadIdx.x;
  if (p >= N) return;
  float s = 0.0f;
  for (int c = 0; c < D; ++c) {
    float v = (float)x[(size_t)p * D + c];
    s += v * v;
  }
  sq[p] = s;
}

// out rows WP-wide bf16: [a(da) | b(db) | zeros]; a float, b bf16 or reversed.
template<typename TA, typename TB>
__global__ void concat_pad_kernel(const TA* __restrict__ a, int da,
                                  const TB* __restrict__ b, int db,
                                  __bf16* __restrict__ out, int N, int WP) {
  int t = blockIdx.x * blockDim.x + threadIdx.x;
  if (t >= N * WP) return;
  int r = t / WP, c = t % WP;
  float v = 0.0f;
  if (c < da) v = (float)a[(size_t)r * da + c];
  else if (c < da + db) v = (float)b[(size_t)r * db + (c - da)];
  out[t] = (__bf16)v;
}

// ---------------------------------------------------------------------------
extern "C" void kernel_launch(void* const* d_in, const int* in_sizes, int n_in,
                              void* d_out, int out_size, void* d_ws, size_t ws_size,
                              hipStream_t stream) {
  (void)in_sizes; (void)n_in; (void)out_size; (void)ws_size;
  const float* ctx_xyz  = (const float*)d_in[0];
  const float* ctx_tok  = (const float*)d_in[1];
  const float* pred_tok = (const float*)d_in[2];
  const float* noise    = (const float*)d_in[3];
  const float* pw[26];
  for (int i = 0; i < 26; ++i) pw[i] = (const float*)d_in[4 + i];
  const float* lx_w1 = pw[0];  const float* lx_b1 = pw[1];
  const float* lx_w2 = pw[2];  const float* lx_b2 = pw[3];
  const float* dc1_w1 = pw[4]; const float* dc1_b1 = pw[5];
  const float* dc1_w2 = pw[6]; const float* dc1_b2 = pw[7];
  const float* dc2_w1 = pw[8]; const float* dc2_b1 = pw[9];
  const float* dc2_w2 = pw[10];const float* dc2_b2 = pw[11];
  const float* cd_w1 = pw[12]; const float* cd_b1 = pw[13];
  const float* cd_w2 = pw[14]; const float* cd_b2 = pw[15];
  const float* fd_w1 = pw[16]; const float* fd_b1 = pw[17];
  const float* fd_w2 = pw[18]; const float* fd_b2 = pw[19];
  const float* fd_w3 = pw[20]; const float* fd_b3 = pw[21];
  const float* rf_w1 = pw[22]; const float* rf_b1 = pw[23];
  const float* rf_w2 = pw[24]; const float* rf_b2 = pw[25];
  const int*   mask_id = (const int*)d_in[30];
  float* out = (float*)d_out;

  char* wsb = (char*)d_ws;
  size_t off = 0;
  auto balloc = [&](size_t bytes) -> void* {
    void* p = (void*)(wsb + off);
    off = (off + bytes + 255) & ~(size_t)255;
    return p;
  };
  auto ballocb = [&](size_t elems) -> __bf16* { return (__bf16*)balloc(elems * 2); };
  auto ballocf = [&](size_t elems) -> float*  { return (float*)balloc(elems * 4); };
  // transposed weights, bf16, K padded to %32
  __bf16* lx_w1t  = ballocb((size_t)384 * 384);
  __bf16* lx_w2t  = ballocb((size_t)3 * 384);
  __bf16* dc1_w1t = ballocb((size_t)192 * 768);
  __bf16* dc1_w2t = ballocb((size_t)192 * 192);
  __bf16* dc2_w1t = ballocb((size_t)96 * 384);
  __bf16* dc2_w2t = ballocb((size_t)96 * 96);
  __bf16* cd_w1t  = ballocb((size_t)48 * 96);
  __bf16* cd_w2t  = ballocb((size_t)3 * 64);    // K 48 -> 64
  __bf16* fd_w1t  = ballocb((size_t)256 * 128); // K 99 -> 128
  __bf16* fd_w2t  = ballocb((size_t)256 * 256);
  __bf16* fd_w3t  = ballocb((size_t)3 * 256);
  __bf16* rf_w1t  = ballocb((size_t)96 * 256);  // split-pad 2x(99->128)
  __bf16* rf_w2t  = ballocb((size_t)3 * 96);
  // activations (bf16 unless residual/final)
  __bf16* ctxtokb = ballocb((size_t)2048 * 384);
  __bf16* predm  = ballocb((size_t)2048 * 384);
  __bf16* hid    = ballocb((size_t)2048 * 384);
  float*  seed   = ballocf((size_t)2048 * 3);
  __bf16* tgttok = ballocb((size_t)8192 * 384);
  float*  txyz1  = ballocf((size_t)8192 * 3);
  float*  txyz2  = ballocf((size_t)8192 * 3);
  __bf16* xyzpad = ballocb((size_t)8192 * 32);
  float*  sqbuf  = ballocf((size_t)8192);
  int*    nbrA   = (int*)balloc((size_t)8192 * 16 * 4);
  int*    nbrB   = (int*)balloc((size_t)8192 * 8 * 4);
  __bf16* feat1  = ballocb((size_t)8192 * 192);
  __bf16* feat2  = ballocb((size_t)8192 * 96);
  __bf16* cdh    = ballocb((size_t)2048 * 64);   // 48 -> stride 64
  __bf16* catbuf = ballocb((size_t)8192 * 128);  // 99 -> stride 128
  __bf16* xgpad  = ballocb((size_t)8192 * 128);  // 99 -> stride 128
  __bf16* fdh1   = ballocb((size_t)8192 * 256);
  __bf16* fdh2   = ballocb((size_t)8192 * 256);
  __bf16* edgebuf = ballocb((size_t)131072 * 192);  // reused across phases

  auto cdiv = [](int a, int b) { return (a + b - 1) / b; };
  dim3 w32(32);

  // ---- transpose (and pad) all weights once, fp32 -> bf16 ----
  struct TP { const float* w; __bf16* wt; int K, N, KP; };
  const TP tps[10] = {
    {lx_w1, lx_w1t, 384, 384, 384}, {lx_w2, lx_w2t, 384, 3, 384},
    {dc1_w2, dc1_w2t, 192, 192, 192}, {dc2_w2, dc2_w2t, 96, 96, 96},
    {cd_w1, cd_w1t, 96, 48, 96}, {cd_w2, cd_w2t, 48, 3, 64},
    {fd_w1, fd_w1t, 99, 256, 128}, {fd_w2, fd_w2t, 256, 256, 256},
    {fd_w3, fd_w3t, 256, 3, 256}, {rf_w2, rf_w2t, 96, 3, 96},
  };
  for (int i = 0; i < 10; ++i)
    transpose_pad_kernel<<<cdiv(tps[i].N * tps[i].KP, 256), 256, 0, stream>>>(
        tps[i].w, tps[i].wt, tps[i].K, tps[i].N, tps[i].KP);
  transpose_edge_pad_kernel<<<cdiv(192 * 768, 256), 256, 0, stream>>>(
      dc1_w1, dc1_w1t, 384, 384, 192);
  transpose_edge_pad_kernel<<<cdiv(96 * 384, 256), 256, 0, stream>>>(
      dc2_w1, dc2_w1t, 192, 192, 96);
  transpose_edge_pad_kernel<<<cdiv(96 * 256, 256), 256, 0, stream>>>(
      rf_w1, rf_w1t, 99, 128, 96);
  f32_to_bf16_kernel<<<cdiv(2048 * 384, 256), 256, 0, stream>>>(
      ctx_tok, ctxtokb, 2048 * 384);

  // ---- seed MLP: pred_tokens[:, mask_id] -> seed_xyz ----
  gather_pred_kernel<<<cdiv(2048 * 384, 256), 256, 0, stream>>>(pred_tok, mask_id, predm);
  gemm_kernel<384, 384, 384, 4, __bf16><<<dim3(128, 6), w32, 0, stream>>>(
      predm, lx_w1t, lx_b1, nullptr, hid, 1, 1.0f);
  gemm_kernel<384, 3, 3, 1, float><<<dim3(128, 1), w32, 0, stream>>>(
      hid, lx_w2t, lx_b2, nullptr, seed, 0, 1.0f);

  // ---- ctx branch: dynconv(2048 x 384) ----
  sqnorm_kernel<<<cdiv(2048, 256), 256, 0, stream>>>(ctxtokb, 2048, 384, sqbuf);
  knn_kernel<384, 16><<<128, w32, 0, stream>>>(ctxtokb, sqbuf, 2048, nbrA);
  gemm_edge_kernel<16, 384, 192, 4><<<dim3(2048, 3), w32, 0, stream>>>(
      ctxtokb, nbrA, dc1_w1t, dc1_b1, edgebuf);
  gemm_groupmax_kernel<16, 192, 192, 4, __bf16><<<dim3(2048, 3), w32, 0, stream>>>(
      edgebuf, dc1_w2t, dc1_b2, nullptr, feat1);
  sqnorm_kernel<<<cdiv(2048, 256), 256, 0, stream>>>(feat1, 2048, 192, sqbuf);
  knn_kernel<192, 8><<<128, w32, 0, stream>>>(feat1, sqbuf, 2048, nbrB);
  gemm_edge_kernel<8, 192, 96, 3><<<dim3(1024, 2), w32, 0, stream>>>(
      feat1, nbrB, dc2_w1t, dc2_b1, edgebuf);
  gemm_groupmax_kernel<8, 96, 96, 3, __bf16><<<dim3(1024, 2), w32, 0, stream>>>(
      edgebuf, dc2_w2t, dc2_b2, nullptr, feat2);
  gemm_kernel<96, 48, 64, 4, __bf16><<<dim3(128, 1), w32, 0, stream>>>(
      feat2, cd_w1t, cd_b1, nullptr, cdh, 1, 1.0f);           // pads cols 48..63
  gemm_kernel<64, 3, 3, 1, float><<<dim3(128, 1), w32, 0, stream>>>(
      cdh, cd_w2t, cd_b2, ctx_xyz, out, 0, 0.05f);            // rows 0..2047

  // ---- tgt branch: dynconv(8192 x 384) ----
  repeat4_kernel<<<cdiv(8192 * 384, 256), 256, 0, stream>>>(predm, tgttok);
  tgtxyz_kernel<<<cdiv(8192 * 3, 256), 256, 0, stream>>>(seed, noise, txyz1);
  sqnorm_kernel<<<cdiv(8192, 256), 256, 0, stream>>>(tgttok, 8192, 384, sqbuf);
  knn_kernel<384, 16><<<512, w32, 0, stream>>>(tgttok, sqbuf, 8192, nbrA);
  gemm_edge_kernel<16, 384, 192, 4><<<dim3(8192, 3), w32, 0, stream>>>(
      tgttok, nbrA, dc1_w1t, dc1_b1, edgebuf);
  gemm_groupmax_kernel<16, 192, 192, 4, __bf16><<<dim3(8192, 3), w32, 0, stream>>>(
      edgebuf, dc1_w2t, dc1_b2, nullptr, feat1);
  sqnorm_kernel<<<cdiv(8192, 256), 256, 0, stream>>>(feat1, 8192, 192, sqbuf);
  knn_kernel<192, 8><<<512, w32, 0, stream>>>(feat1, sqbuf, 8192, nbrB);
  gemm_edge_kernel<8, 192, 96, 3><<<dim3(4096, 2), w32, 0, stream>>>(
      feat1, nbrB, dc2_w1t, dc2_b1, edgebuf);
  gemm_groupmax_kernel<8, 96, 96, 3, __bf16><<<dim3(4096, 2), w32, 0, stream>>>(
      edgebuf, dc2_w2t, dc2_b2, nullptr, feat2);

  // ---- fold MLP: [xyz | feat] (pad 99->128) -> 256 -> 256 -> 3 (+residual) ----
  concat_pad_kernel<<<cdiv(8192 * 128, 256), 256, 0, stream>>>(
      txyz1, 3, feat2, 96, catbuf, 8192, 128);
  gemm_kernel<128, 256, 256, 4, __bf16><<<dim3(512, 4), w32, 0, stream>>>(
      catbuf, fd_w1t, fd_b1, nullptr, fdh1, 1, 1.0f);
  gemm_kernel<256, 256, 256, 4, __bf16><<<dim3(512, 4), w32, 0, stream>>>(
      fdh1, fd_w2t, fd_b2, nullptr, fdh2, 1, 1.0f);
  gemm_kernel<256, 3, 3, 1, float><<<dim3(512, 1), w32, 0, stream>>>(
      fdh2, fd_w3t, fd_b3, txyz1, txyz2, 0, 1.0f);

  // ---- refine: kNN in xyz (padded to 32), edge-conv [feat|xyz] -> 96 -> 3 ----
  concat_pad_kernel<<<cdiv(8192 * 32, 256), 256, 0, stream>>>(
      txyz2, 3, txyz2, 0, xyzpad, 8192, 32);
  sqnorm_kernel<<<cdiv(8192, 256), 256, 0, stream>>>(xyzpad, 8192, 32, sqbuf);
  knn_kernel<32, 16><<<512, w32, 0, stream>>>(xyzpad, sqbuf, 8192, nbrA);
  concat_pad_kernel<<<cdiv(8192 * 128, 256), 256, 0, stream>>>(
      feat2, 96, txyz2, 3, xgpad, 8192, 128);
  gemm_edge_kernel<16, 128, 96, 3><<<dim3(8192, 2), w32, 0, stream>>>(
      xgpad, nbrA, rf_w1t, rf_b1, edgebuf);
  gemm_groupmax_kernel<16, 96, 3, 1, float><<<dim3(8192, 1), w32, 0, stream>>>(
      edgebuf, rf_w2t, rf_b2, txyz2, out + 2048 * 3);  // rows 2048..10239
}
